// TransformerMultiheadAttention_68178310856902
// MI455X (gfx1250) — matrix operand order repaired and checked by reading, not compiled, not run on hardware
//
#include <hip/hip_runtime.h>

// ---------------------------------------------------------------------------
// MI455X (gfx1250) fused multi-head attention forward.
//   B=2, L=2048, D=1024, H=16, HD=64.  wave32, WMMA f32_16x16x32_f16,
//   TDM (tensor_load_to_lds) staging of GEMM A-tiles.
// ---------------------------------------------------------------------------

typedef _Float16 half_t;
typedef _Float16 v16h __attribute__((ext_vector_type(16)));
typedef _Float16 v8h  __attribute__((ext_vector_type(8)));
typedef float    v8f  __attribute__((ext_vector_type(8)));
typedef unsigned int u32x4 __attribute__((ext_vector_type(4)));
typedef int          i32x4 __attribute__((ext_vector_type(4)));
typedef int          i32x8 __attribute__((ext_vector_type(8)));

static constexpr int Bn   = 2;
static constexpr int Lseq = 2048;
static constexpr int Dm   = 1024;
static constexpr int Hh   = 16;
static constexpr int HDd  = 64;

// Padded LDS row stride for the staged A tile: 2048 data bytes + 2x16B pads
// (TDM pad_interval=7 -> pad every 1024 bytes stored; pad_amount=3 -> 4 DWORDs).
static constexpr int LDS_ROW = 2048 + 32;   // 2080 bytes

#define WMMA_F16(A, B, C) \
  __builtin_amdgcn_wmma_f32_16x16x32_f16(false, (A), false, (B), (short)0, (C), false, false)

union AFrag { v16h v; v8h h[2]; };

// ---------------------------------------------------------------------------
// fp32 -> f16 elementwise conversion
// ---------------------------------------------------------------------------
__global__ void cvt_f32_f16(const float* __restrict__ src,
                            half_t* __restrict__ dst, int n) {
  int i = blockIdx.x * blockDim.x + threadIdx.x;
  int stride = gridDim.x * blockDim.x;
  for (; i < n; i += stride) dst[i] = (half_t)src[i];
}

// ---------------------------------------------------------------------------
// TDM: DMA one 16-row x 1024-col f16 tile (row stride 1024) into LDS with
// 16B padding per 1KB (see LDS_ROW). D# built per CDNA5 ISA ch.8.
// This toolchain exposes the 6-arg builtin:
//   (u32x4 group0, i32x8 group1, i32x4 group2, i32x4 group3, i32x8 extra, i32 cpol)
// Groups 2/3 + extra are zero: 2D tensor (VADDR2/3 NULL in the raw encoding).
// ---------------------------------------------------------------------------
#if __has_builtin(__builtin_amdgcn_tensor_load_to_lds)
__device__ __forceinline__ void tdm_load_tile_a(const half_t* gptr, unsigned ldsOff) {
  unsigned long long ga = (unsigned long long)(uintptr_t)gptr;
  u32x4 g0;
  g0[0] = 1u;                                          // count=1 (valid user D#)
  g0[1] = ldsOff;                                      // lds_addr (bytes)
  g0[2] = (unsigned)(ga & 0xFFFFFFFFu);                // global_addr[31:0]
  g0[3] = (unsigned)((ga >> 32) & 0x01FFFFFFu)         // global_addr[56:32]
        | (2u << 30);                                  // type = 2 ("image")
  i32x8 g1;
  g1[0] = (int)((1u << 16)      // data_size = 1 -> 2 bytes
              | (1u << 20)      // pad_enable
              | (7u << 22)      // pad_interval: every 256 DWORDs (1KB)
              | (3u << 25));    // pad_amount: 4 DWORDs (16B)
  g1[1] = (int)(1024u << 16);   // tensor_dim0[15:0] = 1024 (elements)
  g1[2] = (int)(16u << 16);     // tensor_dim1[15:0] = 16
  g1[3] = (int)(1024u << 16);   // tile_dim0 = 1024
  g1[4] = 16;                   // tile_dim1 = 16, tile_dim2 = 0
  g1[5] = 1024;                 // tensor_dim0_stride[31:0] = 1024 (elements)
  g1[6] = 0;
  g1[7] = 0;
  i32x4 z4 = {0, 0, 0, 0};                      // groups 2/3 unused (2D tensor)
  i32x8 z8 = {0, 0, 0, 0, 0, 0, 0, 0};          // extra operand (unused)
  __builtin_amdgcn_tensor_load_to_lds(g0, g1, z4, z4, z8, 0);
}
#endif

// ---------------------------------------------------------------------------
// GEMM: out[m,n] = sum_k A[m,k] * W[n,k]   (A: [4096,1024] f16, W: [1024,1024] f16)
// Block = 8 waves sharing one 16-row A strip (TDM-staged in LDS); each wave
// computes a 16(M) x 64(N) strip with 4 f32 accumulator tiles.
// mode 0: store f16 to [B,H,L,HD]   (q / k projections)
// mode 1: store f16 to [B,H,HD,L]   (v projection, transposed)
// mode 2: store f32 to [M,N] + bias (output projection)
// grid: (2, 256), block: 256 (8 waves)
// ---------------------------------------------------------------------------
__global__ void gemm_xwT(const half_t* __restrict__ A, const half_t* __restrict__ W,
                         half_t* __restrict__ outH, float* __restrict__ outF,
                         const float* __restrict__ bias, int mode)
{
  __shared__ __align__(16) unsigned char ldsA[16 * LDS_ROW];

  const int lane = threadIdx.x & 31;
  const int wave = threadIdx.x >> 5;
  const int l16  = lane & 15;
  const int hi   = lane >> 4;

  const int K  = Dm;
  const int m0 = blockIdx.y * 16;
  const int n0 = (blockIdx.x * 8 + wave) * 64;

  const half_t* aTile = A + (size_t)m0 * K;

#if __has_builtin(__builtin_amdgcn_tensor_load_to_lds)
  if (wave == 0) {
    tdm_load_tile_a(aTile, (unsigned)(uintptr_t)&ldsA[0]);
    __builtin_amdgcn_s_wait_tensorcnt(0);
  }
#else
  // Cooperative fallback: identical padded layout.
  for (int c = threadIdx.x; c < 16 * 128; c += 256) {
    const int row = c >> 7;
    const int p   = (c & 127) * 16;           // 16B chunk offset within row
    const int pp  = p + 16 * (p >> 10);       // fold in LDS padding
    *(v8h*)(&ldsA[row * LDS_ROW + pp]) =
        *(const v8h*)((const unsigned char*)(aTile + (size_t)row * K) + p);
  }
#endif
  __syncthreads();

  v8f acc[4] = {};
  const unsigned char* aRow = &ldsA[l16 * LDS_ROW];

  for (int kb = 0; kb < K; kb += 32) {
    // A fragment (16x32 f16): lane m=l16; halves 0..7 <- K=16*hi+j, 8..15 <- K=16*hi+8+j
    unsigned p0 = (unsigned)(kb * 2 + 32 * hi);
    unsigned p1 = p0 + 16;
    AFrag af;
    af.h[0] = *(const v8h*)(aRow + (p0 + 16 * (p0 >> 10)));
    af.h[1] = *(const v8h*)(aRow + (p1 + 16 * (p1 >> 10)));
#pragma unroll
    for (int t = 0; t < 4; ++t) {
      // B fragment (32x16 f16): lane n=l16 holds W-row n, K = 16*hi + j
      const half_t* wRow = W + (size_t)(n0 + t * 16 + l16) * K + kb + 16 * hi;
      v16h bf = *(const v16h*)wRow;
      acc[t] = WMMA_F16(af.v, bf, acc[t]);
    }
  }

  // C/D layout: lane l, VGPR r  ->  (row = m0 + 8*hi + r, col = n0 + t*16 + l16)
#pragma unroll
  for (int t = 0; t < 4; ++t) {
    const int col = n0 + t * 16 + l16;
#pragma unroll
    for (int r = 0; r < 8; ++r) {
      const int row = m0 + 8 * hi + r;
      const float v = acc[t][r];
      if (mode == 2) {
        outF[(size_t)row * Dm + col] = v + bias[col];
      } else {
        const int b = row >> 11, l = row & (Lseq - 1);   // L = 2048
        const int h = col >> 6,  hd = col & (HDd - 1);
        const size_t idx = (mode == 0)
            ? ((size_t)(b * Hh + h) * Lseq + l) * HDd + hd    // [B,H,L,HD]
            : ((size_t)(b * Hh + h) * HDd + hd) * Lseq + l;   // [B,H,HD,L]
        outH[idx] = (half_t)v;
      }
    }
  }
}

// ---------------------------------------------------------------------------
// Flash attention (causal, online softmax). One wave = 16 query rows.
// grid: (B*H = 32, L/(16*8) = 16), block: 256 (8 waves)
//   qp, kp: [B,H,L,HD] f16;  vT: [B,H,HD,L] f16;  attn out: [B*L, D] f16
// ---------------------------------------------------------------------------
__global__ void flash_attn(const half_t* __restrict__ qp, const half_t* __restrict__ kp,
                           const half_t* __restrict__ vT, half_t* __restrict__ attn)
{
  __shared__ __align__(16) half_t ldsP[8][16][32];   // per-wave P tile (16 q x 32 k)

  const int lane = threadIdx.x & 31;
  const int wave = threadIdx.x >> 5;
  const int l16  = lane & 15;
  const int hi   = lane >> 4;

  const int bh = blockIdx.x;            // b*H + h
  const int b  = bh >> 4;
  const int h  = bh & 15;
  const int qBase = (blockIdx.y * 8 + wave) * 16;

  const half_t* qHead = qp + (size_t)bh * Lseq * HDd;
  const half_t* kHead = kp + (size_t)bh * Lseq * HDd;
  const half_t* vHead = vT + (size_t)bh * HDd * Lseq;

  // Q fragments over hd (K = 0..31 and 32..63)
  AFrag aq0, aq1;
  {
    const half_t* qRow = qHead + (size_t)(qBase + l16) * HDd;
    aq0.h[0] = *(const v8h*)(qRow +      16 * hi);
    aq0.h[1] = *(const v8h*)(qRow + 8  + 16 * hi);
    aq1.h[0] = *(const v8h*)(qRow + 32 + 16 * hi);
    aq1.h[1] = *(const v8h*)(qRow + 40 + 16 * hi);
  }

  v8f O[4] = {};
  float m_i[8], l_i[8];
  const float NEG_INF = -__builtin_inff();
#pragma unroll
  for (int r = 0; r < 8; ++r) { m_i[r] = NEG_INF; l_i[r] = 0.0f; }

  const float scale = 0.125f;                    // 1/sqrt(HD)
  const int nkb = (qBase + 47) >> 5;             // causal: 32-wide K blocks

  for (int kb = 0; kb < nkb; ++kb) {
    const int kBase = kb * 32;

    // ---- S = q @ k^T  (two 16x16 score tiles over this 32-wide k block) ----
    v8f S[2] = {};
#pragma unroll
    for (int kt = 0; kt < 2; ++kt) {
      const half_t* kRow = kHead + (size_t)(kBase + kt * 16 + l16) * HDd;
      v16h bk0 = *(const v16h*)(kRow +      16 * hi);   // hd 0..31
      v16h bk1 = *(const v16h*)(kRow + 32 + 16 * hi);   // hd 32..63
      S[kt] = WMMA_F16(aq0.v, bk0, S[kt]);
      S[kt] = WMMA_F16(aq1.v, bk1, S[kt]);
    }

    // ---- scale, causal mask, online softmax; stage P into LDS ----
    float mNew[8];
#pragma unroll
    for (int r = 0; r < 8; ++r) {
      const int qRowG = qBase + 8 * hi + r;
      float s0 = S[0][r] * scale;
      float s1 = S[1][r] * scale;
      if (kBase + l16      > qRowG) s0 = NEG_INF;
      if (kBase + 16 + l16 > qRowG) s1 = NEG_INF;
      S[0][r] = s0; S[1][r] = s1;
      float mx = fmaxf(s0, s1);
      mx = fmaxf(mx, __shfl_xor(mx, 1, 16));
      mx = fmaxf(mx, __shfl_xor(mx, 2, 16));
      mx = fmaxf(mx, __shfl_xor(mx, 4, 16));
      mx = fmaxf(mx, __shfl_xor(mx, 8, 16));
      mNew[r] = fmaxf(m_i[r], mx);
    }

#pragma unroll
    for (int r = 0; r < 8; ++r) {
      const float p0 = __expf(S[0][r] - mNew[r]);
      const float p1 = __expf(S[1][r] - mNew[r]);
      const float alpha = __expf(m_i[r] - mNew[r]);   // first block: exp(-inf)=0
      float rs = p0 + p1;
      rs += __shfl_xor(rs, 1, 16);
      rs += __shfl_xor(rs, 2, 16);
      rs += __shfl_xor(rs, 4, 16);
      rs += __shfl_xor(rs, 8, 16);
      l_i[r] = l_i[r] * alpha + rs;
      m_i[r] = mNew[r];
#pragma unroll
      for (int t = 0; t < 4; ++t) O[t][r] *= alpha;
      ldsP[wave][8 * hi + r][l16]      = (half_t)p0;
      ldsP[wave][8 * hi + r][16 + l16] = (half_t)p1;
    }

    // same-wave LDS in-order; wait + compiler barrier before re-layout load
    asm volatile("s_wait_dscnt 0" ::: "memory");

    // ---- O += P @ V : P re-read in A-fragment layout, V from transposed ws ----
    AFrag ap;
    ap.h[0] = *(const v8h*)(&ldsP[wave][l16][16 * hi]);
    ap.h[1] = *(const v8h*)(&ldsP[wave][l16][8 + 16 * hi]);
#pragma unroll
    for (int t = 0; t < 4; ++t) {
      const half_t* vRow = vHead + (size_t)(t * 16 + l16) * Lseq + kBase + 16 * hi;
      v16h bv = *(const v16h*)vRow;
      O[t] = WMMA_F16(ap.v, bv, O[t]);
    }
  }

  // ---- normalize and store merged-head output [B*L, D] as f16 ----
#pragma unroll
  for (int r = 0; r < 8; ++r) {
    const float inv = 1.0f / l_i[r];
#pragma unroll
    for (int t = 0; t < 4; ++t) O[t][r] *= inv;
  }
  half_t* aOut = attn + (size_t)b * Lseq * Dm + (size_t)h * HDd;
#pragma unroll
  for (int t = 0; t < 4; ++t)
#pragma unroll
    for (int r = 0; r < 8; ++r)
      aOut[(size_t)(qBase + 8 * hi + r) * Dm + t * 16 + l16] = (half_t)O[t][r];
}

// ---------------------------------------------------------------------------
// Launch
// ---------------------------------------------------------------------------
extern "C" void kernel_launch(void* const* d_in, const int* in_sizes, int n_in,
                              void* d_out, int out_size, void* d_ws, size_t ws_size,
                              hipStream_t stream)
{
  (void)in_sizes; (void)n_in; (void)out_size; (void)ws_size;

  const float* Q  = (const float*)d_in[0];
  const float* K  = (const float*)d_in[1];
  const float* V  = (const float*)d_in[2];
  // d_in[3] = causal mask (implemented analytically)
  const float* wq = (const float*)d_in[4];
  const float* wk = (const float*)d_in[5];
  const float* wv = (const float*)d_in[6];
  const float* wo = (const float*)d_in[7];
  const float* bo = (const float*)d_in[8];
  float* out = (float*)d_out;

  const size_t nX = (size_t)Bn * Lseq * Dm;   // 4,194,304
  const size_t nW = (size_t)Dm * Dm;          // 1,048,576

  char* ws = (char*)d_ws;
  half_t* Qh    = (half_t*)ws;  ws += nX * sizeof(half_t);
  half_t* Kh    = (half_t*)ws;  ws += nX * sizeof(half_t);
  half_t* Vh    = (half_t*)ws;  ws += nX * sizeof(half_t);
  half_t* Wqh   = (half_t*)ws;  ws += nW * sizeof(half_t);
  half_t* Wkh   = (half_t*)ws;  ws += nW * sizeof(half_t);
  half_t* Wvh   = (half_t*)ws;  ws += nW * sizeof(half_t);
  half_t* Woh   = (half_t*)ws;  ws += nW * sizeof(half_t);
  half_t* qpws  = (half_t*)ws;  ws += nX * sizeof(half_t);
  half_t* kpws  = (half_t*)ws;  ws += nX * sizeof(half_t);
  half_t* vTws  = (half_t*)ws;  ws += nX * sizeof(half_t);
  half_t* attnh = (half_t*)ws;  ws += nX * sizeof(half_t);   // ~64 MB total

  // 1) fp32 -> f16 conversions
  cvt_f32_f16<<<4096, 256, 0, stream>>>(Q,  Qh,  (int)nX);
  cvt_f32_f16<<<4096, 256, 0, stream>>>(K,  Kh,  (int)nX);
  cvt_f32_f16<<<4096, 256, 0, stream>>>(V,  Vh,  (int)nX);
  cvt_f32_f16<<<1024, 256, 0, stream>>>(wq, Wqh, (int)nW);
  cvt_f32_f16<<<1024, 256, 0, stream>>>(wk, Wkh, (int)nW);
  cvt_f32_f16<<<1024, 256, 0, stream>>>(wv, Wvh, (int)nW);
  cvt_f32_f16<<<1024, 256, 0, stream>>>(wo, Woh, (int)nW);

  // 2) q/k/v projections (WMMA + TDM A-tile staging)
  dim3 gg(2, 256), gb(256);
  gemm_xwT<<<gg, gb, 0, stream>>>(Qh, Wqh, qpws, nullptr, nullptr, 0);
  gemm_xwT<<<gg, gb, 0, stream>>>(Kh, Wkh, kpws, nullptr, nullptr, 0);
  gemm_xwT<<<gg, gb, 0, stream>>>(Vh, Wvh, vTws, nullptr, nullptr, 1);

  // 3) causal flash attention (WMMA)
  flash_attn<<<dim3(Bn * Hh, Lseq / 128), 256, 0, stream>>>(qpws, kpws, vTws, attnh);

  // 4) output projection + bias (WMMA, fp32 out)
  gemm_xwT<<<gg, gb, 0, stream>>>(attnh, Woh, nullptr, out, bo, 2);
}
